// MaillardSnliModel_1846835937415
// MI455X (gfx1250) — compile-verified, matrix operand order checked by reading, not run on hardware
//
#include <hip/hip_runtime.h>
#include <hip/hip_bf16.h>
#include <math.h>

typedef __attribute__((ext_vector_type(16))) _Float16 v16h;
typedef __attribute__((ext_vector_type(8)))  float    v8f;

#define BSZ  64
#define LLEN 64      // L
#define SSTP 63      // S
#define AMBN 32      // AMB
#define HD   128     // H
#define ED   300     // E
#define GD   640     // 5*H
#define TWOH 256     // 2*H
#define EPAD 320     // E padded to mult of 32 for K
#define MLPN 1024
#define KCAT 512     // 4*H
#define CROWS 127    // L + S
#define EPSV 1e-8f

// ---------------- LDS layout (bytes) ----------------
#define OFF_CHART 0
#define SZ_CHART  (CROWS*TWOH*4)           // 130048
#define OFF_PRE   (OFF_CHART + SZ_CHART)   // 130048
#define SZ_PRE    (32*GD*4)                // 81920 (also reused as reduction scratch)
#define OFF_SCR   (OFF_PRE + SZ_PRE)       // 211968
#define SZ_SCR    (64*EPAD*2)              // 40960 (emb f16 / hh f16 / cV+hV f32)
#define OFF_OPS   (OFF_SCR + SZ_SCR)       // 252928
#define OFF_SV    (OFF_OPS + 64*4)         // [32] softmax weights
#define OFF_MISC  (OFF_SV + 32*4)
#define LDS_BYTES (OFF_MISC + 64)          // 253376 < 320KB

// ---------------- workspace layout (bytes) ----------------
#define WS_UH   0                                   // 640*256 f16
#define WS_WH   (WS_UH + GD*TWOH*2)                 // 640*320 f16
#define WS_AH   (WS_WH + GD*EPAD*2)                 // 1024*512 f16
#define WS_S1   (WS_AH + MLPN*KCAT*2)               // 64*128 f32
#define WS_S2   (WS_S1 + BSZ*HD*4)
#define WS_CONC (WS_S2 + BSZ*HD*4)                  // 64*512 f16

// A/B fragment (16x32 f16, row-major source [row][k]):
//   lane -> row = lane&15 ; k-half = (lane>>4)*8
//   f[0..7]  = K (khalf .. khalf+7),  f[8..15] = K (16+khalf .. 16+khalf+7)
__device__ static inline v16h load_frag16(const _Float16* base, int row0, int ldk,
                                          int k0, int lane) {
  const int r  = lane & 15;
  const int kh = (lane >> 4) << 3;
  const _Float16* p = base + (size_t)(row0 + r) * ldk + k0 + kh;
  v16h f;
#pragma unroll
  for (int j = 0; j < 8; ++j) f[j] = p[j];
#pragma unroll
  for (int j = 0; j < 8; ++j) f[8 + j] = p[16 + j];
  return f;
}

// C/D 16x16 f32: lane -> N = lane&15 ; VGPR r -> M = r + 8*(lane>>4)
__device__ static inline void store_acc_lds(float* dst, int row0, int ldn, int n0,
                                            int lane, v8f acc) {
  const int n  = lane & 15;
  const int mo = (lane >> 4) << 3;
#pragma unroll
  for (int r = 0; r < 8; ++r) dst[(row0 + mo + r) * ldn + n0 + n] = acc[r];
}

// ---------------- prep: fp32 -> fp16 weights ----------------
__global__ void prep_kernel(const float* __restrict__ W, const float* __restrict__ U,
                            const float* __restrict__ A,
                            _Float16* __restrict__ Wh, _Float16* __restrict__ Uh,
                            _Float16* __restrict__ Ah) {
  const int stride = gridDim.x * blockDim.x;
  const int i0 = blockIdx.x * blockDim.x + threadIdx.x;
  for (int t = i0; t < GD * EPAD; t += stride) {
    int g = t / EPAD, k = t - g * EPAD;
    Wh[t] = (_Float16)((k < ED) ? W[g * ED + k] : 0.0f);
  }
  for (int t = i0; t < GD * TWOH; t += stride) Uh[t] = (_Float16)U[t];
  for (int t = i0; t < MLPN * KCAT; t += stride) Ah[t] = (_Float16)A[t];
}

// ---------------- chart recursion: one workgroup per (batch, sentence-set) ----------------
__global__ __launch_bounds__(256)
void chart_kernel(const int* __restrict__ sent1, const int* __restrict__ ops1,
                  const int* __restrict__ oopl1,
                  const int* __restrict__ sent2, const int* __restrict__ ops2,
                  const int* __restrict__ oopl2,
                  const float* __restrict__ bvec, const float* __restrict__ eu,
                  const float* __restrict__ unk,  const float* __restrict__ we,
                  const float* __restrict__ invT,
                  const _Float16* __restrict__ Wh, const _Float16* __restrict__ Uh,
                  float* __restrict__ s1out, float* __restrict__ s2out) {
  extern __shared__ char smem[];
  float*    chart = (float*)(smem + OFF_CHART);     // [127][256]
  float*    pre   = (float*)(smem + OFF_PRE);       // [32][640] (reused as partials)
  _Float16* scrH  = (_Float16*)(smem + OFF_SCR);    // emb[64][320] / hh[32][256]
  float*    scrF  = (float*)(smem + OFF_SCR);       // cV[32][128], hV[32][128]
  int*      opsL  = (int*)(smem + OFF_OPS);         // [64]
  float*    sV    = (float*)(smem + OFF_SV);        // [32]
  float*    misc  = (float*)(smem + OFF_MISC);

  const int b     = blockIdx.x;
  const int which = blockIdx.y;
  const int*  sent = which ? sent2 : sent1;
  const int*  ops  = which ? ops2  : ops1;
  const int*  oopl = which ? oopl2 : oopl1;
  float*      sOut = which ? s2out : s1out;

  const int tid  = threadIdx.x;
  const int lane = tid & 31;
  const int wave = tid >> 5;
  const int mt   = wave & 1;   // constant M-tile parity per wave

  if (tid == 0) {
    float s = 0.f;
    for (int k = 0; k < HD; ++k) s += eu[k] * eu[k];
    misc[0] = fmaxf(sqrtf(s), EPSV);   // ||energy_u||
    misc[1] = invT[0];
  }

  // ===== leaf phase: emb (f16, K padded to 320) =====
  for (int idx = tid; idx < LLEN * EPAD; idx += 256) {
    int l = idx / EPAD, k = idx - l * EPAD;
    float v = 0.f;
    if (k < ED) {
      int tok = sent[b * LLEN + l];
      v = (tok >= 0) ? we[(size_t)tok * ED + k] : unk[k];
    }
    scrH[idx] = (_Float16)v;
  }
  __syncthreads();

  for (int half = 0; half < 2; ++half) {
    // pre[a][g] = emb[half*32+a][:] . W[g][:]   (M=32, N=640, K=320)
    {
      v16h afr[10];
#pragma unroll
      for (int kt = 0; kt < 10; ++kt)
        afr[kt] = load_frag16(scrH, half * 32 + mt * 16, EPAD, kt * 32, lane);
      for (int t = wave; t < 80; t += 8) {
        int n0 = (t >> 1) * 16;
        v8f acc = {};
#pragma unroll
        for (int kt = 0; kt < 10; ++kt)
          acc = __builtin_amdgcn_wmma_f32_16x16x32_f16(
              false, afr[kt], false, load_frag16(Wh, n0, EPAD, kt * 32, lane),
              (short)0, acc, false, false);
        store_acc_lds(pre, mt * 16, GD, n0, lane, acc);
      }
    }
    __syncthreads();
    // gates, ccL = ccR = 0
    for (int idx = tid; idx < 32 * HD; idx += 256) {
      int a = idx >> 7, hx = idx & 127;
      const float* pr = pre + a * GD;
      float gi = 1.f / (1.f + expf(-(pr[hx] + bvec[hx])));
      float go = 1.f / (1.f + expf(-(pr[3 * HD + hx] + bvec[3 * HD + hx])));
      float gu = tanhf(pr[4 * HD + hx] + bvec[4 * HD + hx]);
      float c = gi * gu;
      float h = go * tanhf(c);
      int row = half * 32 + a;
      chart[row * TWOH + hx]      = c;
      chart[row * TWOH + HD + hx] = h;
    }
    __syncthreads();
  }

  // ===== sequential scan, fully LDS-resident =====
  for (int step = 0; step < SSTP; ++step) {
    const int pos = LLEN + step;
    if (tid < 64) opsL[tid] = ops[((size_t)b * SSTP + step) * 64 + tid];
    __syncthreads();

    // hh[a][k] : k<128 -> h of left child, k>=128 -> h of right child
    for (int idx = tid; idx < 32 * TWOH; idx += 256) {
      int a = idx >> 8, k = idx & 255;
      int child = opsL[2 * a + (k >> 7)];
      scrH[idx] = (_Float16)chart[child * TWOH + HD + (k & 127)];
    }
    __syncthreads();

    // pre[a][g] = hh[a][:] . U[g][:]   (M=32, N=640, K=256)
    {
      v16h afr[8];
#pragma unroll
      for (int kt = 0; kt < 8; ++kt)
        afr[kt] = load_frag16(scrH, mt * 16, TWOH, kt * 32, lane);
      for (int t = wave; t < 80; t += 8) {
        int n0 = (t >> 1) * 16;
        __builtin_prefetch(Uh + (size_t)n0 * TWOH, 0, 1);
        v8f acc = {};
#pragma unroll
        for (int kt = 0; kt < 8; ++kt)
          acc = __builtin_amdgcn_wmma_f32_16x16x32_f16(
              false, afr[kt], false, load_frag16(Uh, n0, TWOH, kt * 32, lane),
              (short)0, acc, false, false);
        store_acc_lds(pre, mt * 16, GD, n0, lane, acc);
      }
    }
    __syncthreads();

    // gates -> cV,hV (reuse scratch; hh dead now)
    float* cV = scrF;
    float* hV = scrF + 32 * HD;
    for (int idx = tid; idx < 32 * HD; idx += 256) {
      int a = idx >> 7, hx = idx & 127;
      const float* pr = pre + a * GD;
      float ccL = chart[opsL[2 * a] * TWOH + hx];
      float ccR = chart[opsL[2 * a + 1] * TWOH + hx];
      float gi  = 1.f / (1.f + expf(-(pr[hx] + bvec[hx])));
      float gfL = 1.f / (1.f + expf(-(pr[HD + hx] + bvec[HD + hx])));
      float gfR = 1.f / (1.f + expf(-(pr[2 * HD + hx] + bvec[2 * HD + hx])));
      float go  = 1.f / (1.f + expf(-(pr[3 * HD + hx] + bvec[3 * HD + hx])));
      float gu  = tanhf(pr[4 * HD + hx] + bvec[4 * HD + hx]);
      float c = gfL * ccL + gfR * ccR + gi * gu;
      float h = go * tanhf(c);
      cV[idx] = c;
      hV[idx] = h;
    }
    __syncthreads();

    // cosine energy: 8 threads x 16 elems per candidate; partials in (dead) pre
    {
      int a = tid >> 3, seg = tid & 7;
      const float* hp = hV + a * HD + seg * 16;
      const float* ep = eu + seg * 16;
      float dot = 0.f, nn = 0.f;
#pragma unroll
      for (int k = 0; k < 16; ++k) { float hv = hp[k]; dot += hv * ep[k]; nn += hv * hv; }
      pre[tid]       = dot;
      pre[256 + tid] = nn;
    }
    __syncthreads();
    // finalize energy + softmax entirely in wave 0 (wave32 shuffle reductions)
    if (tid < 32) {
      float dot = 0.f, nn = 0.f;
#pragma unroll
      for (int j = 0; j < 8; ++j) {
        dot += pre[tid * 8 + j];
        nn  += pre[256 + tid * 8 + j];
      }
      float hn = fmaxf(sqrtf(nn), EPSV);
      float e  = (dot / (hn * misc[0])) * misc[1];
      float m = e;
#pragma unroll
      for (int off = 16; off; off >>= 1) m = fmaxf(m, __shfl_xor(m, off, 32));
      float p = expf(e - m);
      float s = p;
#pragma unroll
      for (int off = 16; off; off >>= 1) s += __shfl_xor(s, off, 32);
      sV[tid] = p / s;
    }
    __syncthreads();

    // combine and write chart[pos]
    if (tid < TWOH) {
      const float* src = (tid < HD) ? cV : hV;
      int hx = tid & 127;
      float s = 0.f;
      for (int a = 0; a < 32; ++a) s += sV[a] * src[a * HD + hx];
      chart[pos * TWOH + tid] = s;
    }
    __syncthreads();
  }

  const int last = oopl[b] - 1;
  if (tid < HD) sOut[b * HD + tid] = chart[last * TWOH + HD + tid];
}

// ---------------- concat features -> f16 ----------------
__global__ void concat_kernel(const float* __restrict__ s1, const float* __restrict__ s2,
                              _Float16* __restrict__ concH) {
  int idx = blockIdx.x * blockDim.x + threadIdx.x;
  if (idx >= BSZ * KCAT) return;
  int b = idx >> 9, j = idx & 511;
  int hx = j & 127, region = j >> 7;
  float a = s1[b * HD + hx], c = s2[b * HD + hx];
  float v;
  if (region == 0)      { float d = a - c; v = d * d; }
  else if (region == 1) v = a * c;
  else if (region == 2) v = a;
  else                  v = c;
  concH[idx] = (_Float16)v;
}

// ---------------- final MLP: out[i][j] = relu(A[j][:].conc[i][:] + a[j]) ----------------
// one wave per block; A-fragments (conc rows) hoisted and reused over 4 N-tiles
__global__ __launch_bounds__(32)
void mlp_kernel(const _Float16* __restrict__ concH, const _Float16* __restrict__ Ah,
                const float* __restrict__ avec, float* __restrict__ out) {
  const int lane = threadIdx.x & 31;
  const int i0 = blockIdx.y * 16;   // batch tile
  v16h afr[16];
#pragma unroll
  for (int kt = 0; kt < 16; ++kt)
    afr[kt] = load_frag16(concH, i0, KCAT, kt * 32, lane);
  const int n  = lane & 15;
  const int mo = (lane >> 4) << 3;
  for (int jt = 0; jt < 4; ++jt) {
    const int j0 = blockIdx.x * 64 + jt * 16;   // MLP output tile
    v8f acc = {};
#pragma unroll
    for (int kt = 0; kt < 16; ++kt)
      acc = __builtin_amdgcn_wmma_f32_16x16x32_f16(
          false, afr[kt], false, load_frag16(Ah, j0, KCAT, kt * 32, lane),
          (short)0, acc, false, false);
    const float bj = avec[j0 + n];
#pragma unroll
    for (int r = 0; r < 8; ++r) {
      int i = i0 + mo + r;
      out[(size_t)i * MLPN + j0 + n] = fmaxf(acc[r] + bj, 0.f);
    }
  }
}

extern "C" void kernel_launch(void* const* d_in, const int* in_sizes, int n_in,
                              void* d_out, int out_size, void* d_ws, size_t ws_size,
                              hipStream_t stream) {
  const int*   sent1 = (const int*)d_in[0];
  const int*   ops1  = (const int*)d_in[1];
  const int*   oopl1 = (const int*)d_in[2];
  const int*   sent2 = (const int*)d_in[3];
  const int*   ops2  = (const int*)d_in[4];
  const int*   oopl2 = (const int*)d_in[5];
  const float* W     = (const float*)d_in[6];
  const float* U     = (const float*)d_in[7];
  const float* bvec  = (const float*)d_in[8];
  const float* eu    = (const float*)d_in[9];
  const float* unk   = (const float*)d_in[10];
  const float* we    = (const float*)d_in[11];
  const float* A     = (const float*)d_in[12];
  const float* avec  = (const float*)d_in[13];
  const float* invT  = (const float*)d_in[14];

  char* ws = (char*)d_ws;
  _Float16* Uh    = (_Float16*)(ws + WS_UH);
  _Float16* Wh    = (_Float16*)(ws + WS_WH);
  _Float16* Ah    = (_Float16*)(ws + WS_AH);
  float*    s1    = (float*)(ws + WS_S1);
  float*    s2    = (float*)(ws + WS_S2);
  _Float16* concH = (_Float16*)(ws + WS_CONC);

  prep_kernel<<<512, 256, 0, stream>>>(W, U, A, Wh, Uh, Ah);

  chart_kernel<<<dim3(BSZ, 2), 256, LDS_BYTES, stream>>>(
      sent1, ops1, oopl1, sent2, ops2, oopl2,
      bvec, eu, unk, we, invT, Wh, Uh, s1, s2);

  concat_kernel<<<(BSZ * KCAT + 255) / 256, 256, 0, stream>>>(s1, s2, concH);

  mlp_kernel<<<dim3(MLPN / 64, BSZ / 16), 32, 0, stream>>>(concH, Ah, avec,
                                                           (float*)d_out);
}
